// ScaledDotProductAttention_3393024163920
// MI455X (gfx1250) — matrix operand order repaired and checked by reading, not compiled, not run on hardware
//
#include <hip/hip_runtime.h>

// ---------------------------------------------------------------------------
// MI455X (gfx1250) multi-head attention forward.
// fp32 I/O, bf16 WMMA compute (v_wmma_f32_16x16x32_bf16), fp32 accumulate.
//   1) projection GEMMs write Q,K (bf16 [b,h,q,d]) and V (bf16 [b,h,d,k])
//   2) fused flash attention: async global->LDS double-buffered K/V tiles,
//      permlane16 row reductions, att_map emitted on the fly
//   3) output projection GEMM (fp32 out)
// Workspace: Qp,Kp,Vpt bf16 (3 x 4MB) + Oh fp32 (8MB) = 20MB of d_ws.
// ---------------------------------------------------------------------------

typedef __attribute__((ext_vector_type(16))) __bf16 v16bf;
typedef __attribute__((ext_vector_type(8)))  float  v8f;

__device__ __forceinline__ v16bf frag_ld2(const __bf16* p0, const __bf16* p1) {
  union { uint4 u[2]; v16bf v; } t;
  t.u[0] = *reinterpret_cast<const uint4*>(p0);
  t.u[1] = *reinterpret_cast<const uint4*>(p1);
  return t.v;
}
__device__ __forceinline__ v16bf frag_ld32(const __bf16* p) { return frag_ld2(p, p + 8); }

__device__ __forceinline__ v8f wmma_bf16(v16bf a, v16bf b, v8f c) {
  return __builtin_amdgcn_wmma_f32_16x16x32_bf16(false, a, false, b, (short)0, c,
                                                 false, false);
}

__device__ __forceinline__ void st_bf4(__bf16* dst, float4 v) {
  union { __bf16 h[4]; uint2 u; } t;
  t.h[0] = (__bf16)v.x; t.h[1] = (__bf16)v.y;
  t.h[2] = (__bf16)v.z; t.h[3] = (__bf16)v.w;
  *reinterpret_cast<uint2*>(dst) = t.u;
}

// ---- CDNA5 async global->LDS copy (ASYNCcnt-tracked), 16B per lane ---------
__device__ __forceinline__ void async_cp16(void* lds, const void* gptr) {
  unsigned loff = (unsigned)(size_t)lds;  // low 32 bits of generic = LDS offset
  asm volatile("global_load_async_to_lds_b128 %0, %1, off"
               :: "v"(loff), "v"(gptr) : "memory");
}
__device__ __forceinline__ void wait_async0() {
  asm volatile("s_wait_asynccnt 0x0" ::: "memory");
}

// ---- VALU row reductions over 16-lane groups (v_permlane16_b32) ------------
__device__ __forceinline__ float pl16(float x, unsigned s0, unsigned s1) {
#if __has_builtin(__builtin_amdgcn_permlane16)
  unsigned u = __float_as_uint(x);
  return __uint_as_float(__builtin_amdgcn_permlane16(u, u, s0, s1, false, false));
#else
  (void)s0; (void)s1;
  return x;
#endif
}
__device__ __forceinline__ float rowmax16(float x) {
#if __has_builtin(__builtin_amdgcn_permlane16)
  x = fmaxf(x, pl16(x, 0x67452301u, 0xEFCDAB89u));  // xor 1
  x = fmaxf(x, pl16(x, 0x45670123u, 0xCDEF89ABu));  // xor 2
  x = fmaxf(x, pl16(x, 0x01234567u, 0x89ABCDEFu));  // xor 4
  x = fmaxf(x, pl16(x, 0xFEDCBA98u, 0x76543210u));  // xor 8
#else
  for (int s = 1; s < 16; s <<= 1) x = fmaxf(x, __shfl_xor(x, s, 32));
#endif
  return x;
}
__device__ __forceinline__ float rowsum16(float x) {
#if __has_builtin(__builtin_amdgcn_permlane16)
  x += pl16(x, 0x67452301u, 0xEFCDAB89u);
  x += pl16(x, 0x45670123u, 0xCDEF89ABu);
  x += pl16(x, 0x01234567u, 0x89ABCDEFu);
  x += pl16(x, 0xFEDCBA98u, 0x76543210u);
#else
  for (int s = 1; s < 16; s <<= 1) x += __shfl_xor(x, s, 32);
#endif
  return x;
}

// ---------------------------------------------------------------------------
// GEMM: C[2048,1024] = A[2048,1024]*B[1024,1024] + bias
// MODE 0: fp32 row-major.  MODE 1: bf16 [b,h,q,d].  MODE 2: bf16 [b,h,d,q].
// Block 256 thr (8 waves), tile 64(M) x 64(N), k-tile 32; wave = 16x32.
// ---------------------------------------------------------------------------
template <int MODE>
__global__ __launch_bounds__(256) void gemm_bias_kernel(
    const float* __restrict__ A, const float* __restrict__ B,
    const float* __restrict__ bias, void* __restrict__ Cv) {
  constexpr int N = 1024, K = 1024;
  constexpr int LDA = 40, LDB = 40;
  __shared__ __bf16 As[64 * LDA];   // [m][k]
  __shared__ __bf16 Bs[64 * LDB];   // [n][k] (transposed)

  const int tid = threadIdx.x;
  const int wave = tid >> 5, lane = tid & 31;
  const int half = lane >> 4, l16 = lane & 15;
  const int m0 = blockIdx.x * 64, n0 = blockIdx.y * 64;
  const int msub = (wave & 3) * 16, nsub = (wave >> 2) * 32;

  v8f acc0 = {}, acc1 = {};

  for (int k0 = 0; k0 < K; k0 += 32) {
#pragma unroll
    for (int i = 0; i < 2; ++i) {  // A tile 64x32 as float4
      int c = tid + 256 * i;
      int m = c >> 3, f = c & 7;
      float4 a4 = *reinterpret_cast<const float4*>(&A[(size_t)(m0 + m) * K + k0 + f * 4]);
      st_bf4(&As[m * LDA + f * 4], a4);
    }
#pragma unroll
    for (int i = 0; i < 2; ++i) {  // B tile 32x64 -> transposed scatter
      int c = tid + 256 * i;
      int k = c >> 4, nf = c & 15;
      float4 b4 = *reinterpret_cast<const float4*>(&B[(size_t)(k0 + k) * N + n0 + nf * 4]);
      Bs[(nf * 4 + 0) * LDB + k] = (__bf16)b4.x;
      Bs[(nf * 4 + 1) * LDB + k] = (__bf16)b4.y;
      Bs[(nf * 4 + 2) * LDB + k] = (__bf16)b4.z;
      Bs[(nf * 4 + 3) * LDB + k] = (__bf16)b4.w;
    }
    __syncthreads();

    const __bf16* ar = As + (msub + l16) * LDA;
    v16bf af = frag_ld32(ar + half * 16);
    v16bf bf0 = frag_ld32(Bs + (nsub + l16) * LDB + half * 16);
    v16bf bf1 = frag_ld32(Bs + (nsub + 16 + l16) * LDB + half * 16);
    acc0 = wmma_bf16(af, bf0, acc0);
    acc1 = wmma_bf16(af, bf1, acc1);
    __syncthreads();
  }

  const int col0 = n0 + nsub + l16, col1 = col0 + 16;
  const float b0 = bias[col0], b1 = bias[col1];
#pragma unroll
  for (int r = 0; r < 8; ++r) {
    const int row = m0 + msub + r + 8 * half;
    const float v0 = acc0[r] + b0, v1 = acc1[r] + b1;
    const int b = row >> 10, q = row & 1023;
    if (MODE == 0) {
      float* C = (float*)Cv;
      C[(size_t)row * N + col0] = v0;
      C[(size_t)row * N + col1] = v1;
    } else if (MODE == 1) {  // bf16 [b,h,q,d]
      __bf16* C = (__bf16*)Cv;
      C[(((size_t)(b * 16 + (col0 >> 6)) * 1024 + q) << 6) + (col0 & 63)] = (__bf16)v0;
      C[(((size_t)(b * 16 + (col1 >> 6)) * 1024 + q) << 6) + (col1 & 63)] = (__bf16)v1;
    } else {                 // bf16 [b,h,d,q]
      __bf16* C = (__bf16*)Cv;
      C[((size_t)(b * 16 + (col0 >> 6)) * 64 + (col0 & 63)) * 1024 + q] = (__bf16)v0;
      C[((size_t)(b * 16 + (col1 >> 6)) * 64 + (col1 & 63)) * 1024 + q] = (__bf16)v1;
    }
  }
}

// ---------------------------------------------------------------------------
// Fused flash attention + att_map emission.
// Grid (nq/128, b*H); 8 waves x 16 q-rows. Double-buffered async K/V tiles.
// ---------------------------------------------------------------------------
__global__ __launch_bounds__(256) void attn_kernel(
    const __bf16* __restrict__ Qp, const __bf16* __restrict__ Kp,
    const __bf16* __restrict__ Vpt, const int* __restrict__ mask,
    const float* __restrict__ wts, float* __restrict__ amap,
    float* __restrict__ Oh) {
  constexpr int NQ = 1024, NK = 1024, D = 64, H = 16;
  constexpr int LDK = 72, LDV = 40, LDP = 40;
  constexpr float SCALE = 0.125f;
  constexpr float NEGINF = -1.0e30f;

  __shared__ __bf16 KtB[2][32 * LDK];      // [k][d]
  __shared__ __bf16 VtB[2][64 * LDV];      // [d][k]
  __shared__ __bf16 Pl[8 * 16 * LDP];      // per-wave P tiles

  const int tid = threadIdx.x;
  const int wave = tid >> 5, lane = tid & 31;
  const int half = lane >> 4, l16 = lane & 15;
  const int bh = blockIdx.y;
  const int q0 = blockIdx.x * 128 + wave * 16;

  // per-thread async staging coordinates (16B chunks)
  const int kr = tid >> 3, kc = tid & 7;   // K tile: 32 rows x 8 chunks
  const int vd = tid >> 2, vc = tid & 3;   // V tile: 64 rows x 4 chunks
  const __bf16* gK = Kp + ((size_t)bh * NK + kr) * D + kc * 8;
  const __bf16* gV = Vpt + ((size_t)bh * D + vd) * NK + vc * 8;
  __bf16* lK = (__bf16*)&KtB[0][kr * LDK + kc * 8];
  __bf16* lV = (__bf16*)&VtB[0][vd * LDV + vc * 8];
  constexpr size_t KBUF = 32 * LDK, VBUF = 64 * LDV;

  // Q fragments straight from global bf16 (row = lane, A layout)
  const __bf16* qrow = Qp + ((size_t)bh * NQ + q0 + l16) * D;
  v16bf qf[2];
#pragma unroll
  for (int di = 0; di < 2; ++di)
    qf[di] = frag_ld32(qrow + di * 32 + half * 16);

  // per-lane bases for mask/weights/att_map (r*NK folds into immediates)
  const size_t rowbase = ((size_t)bh * NQ + q0 + 8 * half) * NK + l16;
  const int* mb = mask + rowbase;
  const float* wb = wts + rowbase;
  float* ab = amap + rowbase;

  v8f o0 = {}, o1 = {}, o2 = {}, o3 = {};
  float mrow[8], lrow[8];
#pragma unroll
  for (int r = 0; r < 8; ++r) { mrow[r] = -3.0e38f; lrow[r] = 0.f; }

  __bf16* Pw = Pl + wave * 16 * LDP;

  // prime buffer 0
  async_cp16(lK, gK);
  async_cp16(lV, gV);
  wait_async0();
  __syncthreads();

  for (int k0 = 0; k0 < NK; k0 += 32) {
    const int cur = (k0 >> 5) & 1;
    if (k0 + 32 < NK) {  // prefetch next tile into other buffer
      async_cp16(lK + (cur ^ 1) * KBUF, gK + (size_t)(k0 + 32) * D);
      async_cp16(lV + (cur ^ 1) * VBUF, gV + (k0 + 32));
    }
    const __bf16* Kt = KtB[cur];
    const __bf16* Vt = VtB[cur];

    // S = Q K^T : 16(q) x 32(k) in two 16x16 halves, K-dim = d (2 steps)
    v8f s0 = {}, s1 = {};
#pragma unroll
    for (int di = 0; di < 2; ++di) {
      const int d0 = di * 32;
      s0 = wmma_bf16(qf[di], frag_ld32(Kt + (0 + l16) * LDK + d0 + half * 16), s0);
      s1 = wmma_bf16(qf[di], frag_ld32(Kt + (16 + l16) * LDK + d0 + half * 16), s1);
    }

    // att_map emit + weights/mask + online softmax
#pragma unroll
    for (int r = 0; r < 8; ++r) {
      const size_t i0 = (size_t)r * NK + k0;
      const float a0 = s0[r] * SCALE, a1 = s1[r] * SCALE;
      ab[i0] = a0;
      ab[i0 + 16] = a1;
      float w0 = (mb[i0] == 0) ? NEGINF : a0 * wb[i0];
      float w1 = (mb[i0 + 16] == 0) ? NEGINF : a1 * wb[i0 + 16];

      const float tmax = rowmax16(fmaxf(w0, w1));
      const float mnew = fmaxf(mrow[r], tmax);
      const float alpha = __expf(mrow[r] - mnew);
      const float p0 = __expf(w0 - mnew);
      const float p1 = __expf(w1 - mnew);
      const float rs = rowsum16(p0 + p1);
      lrow[r] = lrow[r] * alpha + rs;
      mrow[r] = mnew;

      const int prow = r + 8 * half;
      Pw[prow * LDP + l16] = (__bf16)p0;
      Pw[prow * LDP + 16 + l16] = (__bf16)p1;

      o0[r] *= alpha; o1[r] *= alpha; o2[r] *= alpha; o3[r] *= alpha;
    }

    // P(16x32) @ V(32x64)
    const __bf16* pr = Pw + l16 * LDP;
    v16bf pf = frag_ld32(pr + half * 16);
    o0 = wmma_bf16(pf, frag_ld32(Vt + (0 * 16 + l16) * LDV + half * 16), o0);
    o1 = wmma_bf16(pf, frag_ld32(Vt + (1 * 16 + l16) * LDV + half * 16), o1);
    o2 = wmma_bf16(pf, frag_ld32(Vt + (2 * 16 + l16) * LDV + half * 16), o2);
    o3 = wmma_bf16(pf, frag_ld32(Vt + (3 * 16 + l16) * LDV + half * 16), o3);

    wait_async0();   // next tile landed
    __syncthreads(); // everyone done reading cur / writing next
  }

  // normalize, write head output [b][q][h][d] == [2048,1024] fp32
  const int b = bh >> 4, h = bh & 15;
#pragma unroll
  for (int r = 0; r < 8; ++r) {
    const int q = q0 + r + 8 * half;
    const float inv = 1.0f / lrow[r];
    const size_t ob = (((size_t)b * NQ + q) * H + h) * D;
    Oh[ob + 0 * 16 + l16] = o0[r] * inv;
    Oh[ob + 1 * 16 + l16] = o1[r] * inv;
    Oh[ob + 2 * 16 + l16] = o2[r] * inv;
    Oh[ob + 3 * 16 + l16] = o3[r] * inv;
  }
}

// ---------------------------------------------------------------------------
extern "C" void kernel_launch(void* const* d_in, const int* in_sizes, int n_in,
                              void* d_out, int out_size, void* d_ws, size_t ws_size,
                              hipStream_t stream) {
  const float* queries = (const float*)d_in[0];
  const float* keys    = (const float*)d_in[1];
  const float* values  = (const float*)d_in[2];
  const int*   amask   = (const int*)d_in[3];
  const float* aw      = (const float*)d_in[4];
  const float* Wq = (const float*)d_in[5];  const float* bq = (const float*)d_in[6];
  const float* Wk = (const float*)d_in[7];  const float* bk = (const float*)d_in[8];
  const float* Wv = (const float*)d_in[9];  const float* bv = (const float*)d_in[10];
  const float* Wo = (const float*)d_in[11]; const float* bo = (const float*)d_in[12];

  float* out  = (float*)d_out;                    // [2,1024,1024]
  float* amap = out + (size_t)2 * 1024 * 1024;    // [2,16,1024,1024]

  const size_t HSZ = (size_t)2 * 16 * 1024 * 64;  // per-tensor elements
  __bf16* Qp  = (__bf16*)d_ws;
  __bf16* Kp  = Qp + HSZ;
  __bf16* Vpt = Kp + HSZ;
  float*  Oh  = (float*)(Vpt + HSZ);

  dim3 blk(256);
  dim3 ggemm(32, 16);  // (2048/64, 1024/64)
  gemm_bias_kernel<1><<<ggemm, blk, 0, stream>>>(queries, Wq, bq, Qp);
  gemm_bias_kernel<1><<<ggemm, blk, 0, stream>>>(keys,    Wk, bk, Kp);
  gemm_bias_kernel<2><<<ggemm, blk, 0, stream>>>(values,  Wv, bv, Vpt);

  dim3 gattn(8, 32);   // (1024/128, b*H)
  attn_kernel<<<gattn, blk, 0, stream>>>(Qp, Kp, Vpt, amask, aw, amap, Oh);

  gemm_bias_kernel<0><<<ggemm, blk, 0, stream>>>(Oh, Wo, bo, out);
}
